// MultiHeadAttention_31052613550603
// MI455X (gfx1250) — matrix operand order repaired and checked by reading, not compiled, not run on hardware
//
#include <hip/hip_runtime.h>

// ---------------------------------------------------------------------------
// MI455X (gfx1250) multi-head attention forward.
// All GEMMs + attention matmuls via v_wmma_f32_16x16x32_bf16 (fp32 accum).
// Weights pre-transposed so every global->LDS tile move is a plain b128 copy,
// which lets us use GLOBAL_LOAD_ASYNC_TO_LDS_B128 double-buffering when the
// toolchain exposes the builtin (sync b128 fallback otherwise).
// D_MODEL=1024, H=16, D_K=D_V=64, B=2, S=2048 -> rows = 4096.
// ---------------------------------------------------------------------------

typedef __bf16 bf16;
typedef __attribute__((ext_vector_type(8)))  __bf16 bf16x8;
typedef __attribute__((ext_vector_type(16))) __bf16 bf16x16;
typedef __attribute__((ext_vector_type(8)))  float  v8f;
typedef __attribute__((ext_vector_type(4)))  int    v4i;

#if __has_builtin(__builtin_amdgcn_global_load_async_to_lds_b128)
#define HAVE_ASYNC_LDS 1
#else
#define HAVE_ASYNC_LDS 0
#endif

// 16-byte global -> LDS copy (async when available).
static __device__ __forceinline__ void copy16_g2l(const bf16* g, bf16* l) {
#if HAVE_ASYNC_LDS
  __builtin_amdgcn_global_load_async_to_lds_b128(
      (__attribute__((address_space(1))) v4i*)g,
      (__attribute__((address_space(3))) v4i*)l, 0, 0);
#else
  *(bf16x8*)l = *(const bf16x8*)g;
#endif
}
static __device__ __forceinline__ void async_wait_all() {
#if HAVE_ASYNC_LDS
  asm volatile("s_wait_asynccnt 0x0" ::: "memory");
#endif
}

static __device__ __forceinline__ bf16x16 joinv(bf16x8 lo, bf16x8 hi) {
  return __builtin_shufflevector(lo, hi, 0,1,2,3,4,5,6,7,8,9,10,11,12,13,14,15);
}
// B-fragment: lane = N; lanes 0-15 hold K=0..15, lanes 16-31 K=16..31;
// 16 contiguous bf16 per lane starting at p (caller applies half offset).
static __device__ __forceinline__ bf16x16 bfrag(const bf16* p) {
  return joinv(*(const bf16x8*)p, *(const bf16x8*)(p + 8));
}
// A-fragment (16-bit A 16x32): lane = M; lanes 0-15: K=k0..k0+15,
// lanes 16-31: K=k0+16..k0+31. Two contiguous 16B chunks.
static __device__ __forceinline__ bf16x16 afrag(const bf16* row, int hf, int k0) {
  return joinv(*(const bf16x8*)(row + k0 + hf * 16),
               *(const bf16x8*)(row + k0 + hf * 16 + 8));
}
static __device__ __forceinline__ v8f wmma_bf16(bf16x16 a, bf16x16 b, v8f c) {
  return __builtin_amdgcn_wmma_f32_16x16x32_bf16(false, a, false, b, (short)0, c,
                                                 false, false);
}

// ---------------------------------------------------------------------------
// Stage 0: conversions / repacks (weights stored TRANSPOSED: BT[n][k])
// ---------------------------------------------------------------------------
__global__ __launch_bounds__(256) void cvt_f32_bf16(const float* __restrict__ in,
                                                    bf16* __restrict__ out, int n) {
  int i = blockIdx.x * 256 + threadIdx.x;
  if (i < n) out[i] = (bf16)in[i];
}

// Wq/Wk/Wv: [16][1024][64] fp32 -> BT[1024][1024] bf16, BT[n][k] with n=h*64+d, k=m
__global__ __launch_bounds__(256) void repack_wT(const float* __restrict__ W,
                                                 bf16* __restrict__ out) {
  int i = blockIdx.x * 256 + threadIdx.x;   // over 1024*1024, i = n*1024 + k
  int n = i >> 10, k = i & 1023;
  int h = n >> 6, d = n & 63;
  out[i] = (bf16)W[(h * 1024 + k) * 64 + d];
}

// Wo: [1024][1024] fp32 row-major [k][n] -> BT[1024][1024] bf16 [n][k]
__global__ __launch_bounds__(256) void transpose_wT(const float* __restrict__ W,
                                                    bf16* __restrict__ out) {
  int i = blockIdx.x * 256 + threadIdx.x;   // i = n*1024 + k
  int n = i >> 10, k = i & 1023;
  out[i] = (bf16)W[k * 1024 + n];
}

// ---------------------------------------------------------------------------
// Stage 1/3: tiled WMMA GEMM  C[4096,1024] = A[4096,1024] x BT^T + bias
// mode 0: fp32 out, row-major [M][N]            (output projection)
// mode 1: bf16 out, [b][h][s][64]               (Q/K projections)
// mode 2: bf16 out, [b*h][64 d][2048 s]         (V projection, pre-transposed)
// WG = 256 thr (8 waves), tile 128x128, KB=32, double-buffered LDS.
// ---------------------------------------------------------------------------
__global__ __launch_bounds__(256) void gemm_bf16_wmma(
    const bf16* __restrict__ A, const bf16* __restrict__ BT,
    const float* __restrict__ bias, float* __restrict__ outF,
    bf16* __restrict__ outB, int mode) {
  __shared__ alignas(16) bf16 As[2][128][32];
  __shared__ alignas(16) bf16 Bs[2][128][32];   // Bs[buf][n][k] (BT rows)

  const int tid  = threadIdx.x;
  const int wv   = tid >> 5, lane = tid & 31;
  const int hf   = lane >> 4, ln = lane & 15;
  const int wm   = wv & 3, wn = wv >> 2;
  const int n0   = blockIdx.x * 128;
  const int m0   = blockIdx.y * 128;

  // cooperative copy coords: 256 thr x 32B = 8KB tile
  const int cm = tid >> 1, ck = (tid & 1) * 16;
  const bf16* gA = A  + (size_t)(m0 + cm) * 1024 + ck;
  const bf16* gB = BT + (size_t)(n0 + cm) * 1024 + ck;

  v8f acc[2][4];
#pragma unroll
  for (int f = 0; f < 2; ++f)
#pragma unroll
    for (int t = 0; t < 4; ++t) acc[f][t] = (v8f){0,0,0,0,0,0,0,0};

  // prologue: stage K-slice 0 into buffer 0
  copy16_g2l(gA,     &As[0][cm][ck]);
  copy16_g2l(gA + 8, &As[0][cm][ck + 8]);
  copy16_g2l(gB,     &Bs[0][cm][ck]);
  copy16_g2l(gB + 8, &Bs[0][cm][ck + 8]);
  async_wait_all();
  __syncthreads();

  for (int i = 0; i < 32; ++i) {
    const int cur = i & 1;
    if (i + 1 < 32) {               // stage next K-slice into the other buffer
      const bf16* na = gA + (i + 1) * 32;
      const bf16* nb = gB + (i + 1) * 32;
      copy16_g2l(na,     &As[cur ^ 1][cm][ck]);
      copy16_g2l(na + 8, &As[cur ^ 1][cm][ck + 8]);
      copy16_g2l(nb,     &Bs[cur ^ 1][cm][ck]);
      copy16_g2l(nb + 8, &Bs[cur ^ 1][cm][ck + 8]);
#if !HAVE_ASYNC_LDS
    } else {
      // keep instruction stream uniform on sync fallback; prefetch epilogue bias
      __builtin_prefetch(bias + n0, 0, 1);
#endif
    }
    bf16x16 af[2], bfv[4];
#pragma unroll
    for (int f = 0; f < 2; ++f)
      af[f] = afrag(&As[cur][wm * 32 + f * 16 + ln][0], hf, 0);
#pragma unroll
    for (int t = 0; t < 4; ++t)
      bfv[t] = bfrag(&Bs[cur][wn * 64 + t * 16 + ln][hf * 16]);
#pragma unroll
    for (int f = 0; f < 2; ++f)
#pragma unroll
      for (int t = 0; t < 4; ++t)
        acc[f][t] = wmma_bf16(af[f], bfv[t], acc[f][t]);
    async_wait_all();
    __syncthreads();
  }

  // epilogue.  D layout: VGPR r, lanes0-15: M=r, lanes16-31: M=r+8
#pragma unroll
  for (int f = 0; f < 2; ++f) {
#pragma unroll
    for (int r = 0; r < 8; ++r) {
      int M = m0 + wm * 32 + f * 16 + hf * 8 + r;
#pragma unroll
      for (int t = 0; t < 4; ++t) {
        int N = n0 + wn * 64 + t * 16 + ln;
        float v = acc[f][t][r] + bias[N];
        if (mode == 0) {
          outF[(size_t)M * 1024 + N] = v;
        } else {
          int bI = M >> 11, s = M & 2047;
          int hh = N >> 6,  d = N & 63;
          if (mode == 1)
            outB[(((size_t)bI * 16 + hh) * 2048 + s) * 64 + d] = (bf16)v;
          else  // mode 2: V transposed per head [bh][d][s]
            outB[(((size_t)bI * 16 + hh) * 64 + d) * 2048 + s] = (bf16)v;
        }
      }
    }
  }
}

// ---------------------------------------------------------------------------
// Stage 2: flash attention.  Grid (32 qblocks, 32 b*h), 128 thr = 4 waves.
// Wave owns 16 query rows; 64 key blocks of 32, double-buffered K/V tiles.
// softmax tracked in base-2 domain (v_exp_f32 native).
// ---------------------------------------------------------------------------
__global__ __launch_bounds__(128) void attn_wmma(
    const bf16* __restrict__ Q, const bf16* __restrict__ K,
    const bf16* __restrict__ VT,            // [bh][64 d][2048 s]
    bf16* __restrict__ Head) {
  __shared__ alignas(16) bf16 Kt[2][32][64];    // [key][d]  (B layout [n][k])
  __shared__ alignas(16) bf16 Vt[2][64][32];    // [d][key]  (B layout for P*V)
  __shared__ alignas(16) bf16 Pl[4][16][32];    // per-wave P staging [M][key]

  const int tid  = threadIdx.x;
  const int w    = tid >> 5;
  const int lane = tid & 31;
  const int hf   = lane >> 4, ln = lane & 15;
  const int bh   = blockIdx.y;
  const int b    = bh >> 4, h = bh & 15;
  const int qb   = blockIdx.x;

  const bf16* Qb  = Q  + (size_t)bh * 2048 * 64;
  const bf16* Kb  = K  + (size_t)bh * 2048 * 64;
  const bf16* VTb = VT + (size_t)bh * 64 * 2048;

  const int q0 = qb * 64 + w * 16;
  const bf16* qrow = Qb + (size_t)(q0 + ln) * 64;
  bf16x16 qf0 = afrag(qrow, hf, 0);
  bf16x16 qf1 = afrag(qrow, hf, 32);

  float mrow[8], lrow[8];
  v8f acc[4];
#pragma unroll
  for (int r = 0; r < 8; ++r) { mrow[r] = -__builtin_inff(); lrow[r] = 0.f; }
#pragma unroll
  for (int t = 0; t < 4; ++t) acc[t] = (v8f){0,0,0,0,0,0,0,0};

  // scores scaled into log2 domain: s2 = s * (1/sqrt(64)) * log2(e)
  const float scale2 = 0.125f * 1.44269504088896340736f;

  // coop copy coords
  const int kkey = tid >> 2, kdc = (tid & 3) * 16;   // K tile: 32 x 64
  const int vd   = tid >> 1, vkc = (tid & 1) * 16;   // V tile: 64 x 32

  // prologue: stage key-block 0 into buffer 0
  {
    const bf16* gk = Kb + (size_t)kkey * 64 + kdc;
    copy16_g2l(gk,     &Kt[0][kkey][kdc]);
    copy16_g2l(gk + 8, &Kt[0][kkey][kdc + 8]);
    const bf16* gv = VTb + (size_t)vd * 2048 + vkc;
    copy16_g2l(gv,     &Vt[0][vd][vkc]);
    copy16_g2l(gv + 8, &Vt[0][vd][vkc + 8]);
  }
  async_wait_all();
  __syncthreads();

  for (int kb = 0; kb < 64; ++kb) {
    const int cur = kb & 1;
    if (kb + 1 < 64) {            // stage next key block into other buffer
      const bf16* gk = Kb + (size_t)((kb + 1) * 32 + kkey) * 64 + kdc;
      copy16_g2l(gk,     &Kt[cur ^ 1][kkey][kdc]);
      copy16_g2l(gk + 8, &Kt[cur ^ 1][kkey][kdc + 8]);
      const bf16* gv = VTb + (size_t)vd * 2048 + (kb + 1) * 32 + vkc;
      copy16_g2l(gv,     &Vt[cur ^ 1][vd][vkc]);
      copy16_g2l(gv + 8, &Vt[cur ^ 1][vd][vkc + 8]);
    }

    // scores S[16q x 32k] = Q(16x64) * K^T : two 16x16 D tiles, 4 WMMAs
    v8f s0 = (v8f){0,0,0,0,0,0,0,0}, s1 = s0;
    s0 = wmma_bf16(qf0, bfrag(&Kt[cur][ln][hf * 16]),           s0);
    s0 = wmma_bf16(qf1, bfrag(&Kt[cur][ln][32 + hf * 16]),      s0);
    s1 = wmma_bf16(qf0, bfrag(&Kt[cur][16 + ln][hf * 16]),      s1);
    s1 = wmma_bf16(qf1, bfrag(&Kt[cur][16 + ln][32 + hf * 16]), s1);

    // online softmax (base-2). Row = 16 lanes of one half, per VGPR r.
#pragma unroll
    for (int r = 0; r < 8; ++r) {
      float a0 = s0[r] * scale2, a1 = s1[r] * scale2;
      float mv = fmaxf(a0, a1);
      mv = fmaxf(mv, __shfl_xor(mv, 1, 16));
      mv = fmaxf(mv, __shfl_xor(mv, 2, 16));
      mv = fmaxf(mv, __shfl_xor(mv, 4, 16));
      mv = fmaxf(mv, __shfl_xor(mv, 8, 16));
      float mnew  = fmaxf(mrow[r], mv);
      float alpha = exp2f(mrow[r] - mnew);
      float p0 = exp2f(a0 - mnew);
      float p1 = exp2f(a1 - mnew);
      float sum = p0 + p1;
      sum += __shfl_xor(sum, 1, 16);
      sum += __shfl_xor(sum, 2, 16);
      sum += __shfl_xor(sum, 4, 16);
      sum += __shfl_xor(sum, 8, 16);
      lrow[r] = lrow[r] * alpha + sum;
      mrow[r] = mnew;
#pragma unroll
      for (int t = 0; t < 4; ++t) acc[t][r] *= alpha;
      // stage P in [M][key] (D->A layout change); same-wave DS ops are
      // in order, so no barrier needed between these stores and the reads.
      Pl[w][hf * 8 + r][ln]      = (bf16)p0;
      Pl[w][hf * 8 + r][16 + ln] = (bf16)p1;
    }

    // O += P(16x32) * V(32x64): 4 WMMAs
    bf16x16 pf = afrag(&Pl[w][ln][0], hf, 0);
#pragma unroll
    for (int t = 0; t < 4; ++t)
      acc[t] = wmma_bf16(pf, bfrag(&Vt[cur][t * 16 + ln][hf * 16]), acc[t]);

    async_wait_all();
    __syncthreads();
  }

  // epilogue: normalize, store concat layout [b][s][h*64+d] (bf16)
#pragma unroll
  for (int r = 0; r < 8; ++r) {
    float inv = 1.0f / lrow[r];
    int M = hf * 8 + r;
    size_t row = (size_t)(b * 2048 + q0 + M) * 1024;
#pragma unroll
    for (int t = 0; t < 4; ++t)
      Head[row + h * 64 + t * 16 + ln] = (bf16)(acc[t][r] * inv);
  }
}

// ---------------------------------------------------------------------------
// Host-side orchestration
// ---------------------------------------------------------------------------
extern "C" void kernel_launch(void* const* d_in, const int* in_sizes, int n_in,
                              void* d_out, int out_size, void* d_ws, size_t ws_size,
                              hipStream_t stream) {
  (void)in_sizes; (void)n_in; (void)out_size; (void)ws_size;
  const float* emb = (const float*)d_in[0];
  const float* Wq  = (const float*)d_in[1];
  const float* bq  = (const float*)d_in[2];
  const float* Wk  = (const float*)d_in[3];
  const float* bk  = (const float*)d_in[4];
  const float* Wv  = (const float*)d_in[5];
  const float* bv  = (const float*)d_in[6];
  const float* Wo  = (const float*)d_in[7];
  const float* bo  = (const float*)d_in[8];
  float* out = (float*)d_out;

  const size_t NX = (size_t)4096 * 1024;    // activations / head rows
  const size_t NW = (size_t)1024 * 1024;    // weight matrices
  char* ws = (char*)d_ws;
  bf16* Xbf  = (bf16*)ws;  ws += NX * sizeof(bf16);
  bf16* WqT  = (bf16*)ws;  ws += NW * sizeof(bf16);
  bf16* WkT  = (bf16*)ws;  ws += NW * sizeof(bf16);
  bf16* WvT  = (bf16*)ws;  ws += NW * sizeof(bf16);
  bf16* WoT  = (bf16*)ws;  ws += NW * sizeof(bf16);
  bf16* Qm   = (bf16*)ws;  ws += NX * sizeof(bf16);
  bf16* Km   = (bf16*)ws;  ws += NX * sizeof(bf16);
  bf16* Vtm  = (bf16*)ws;  ws += NX * sizeof(bf16);
  bf16* Head = Xbf;   // safe reuse: Xbf last read by V-GEMM; Head written after

  cvt_f32_bf16<<<(int)(NX / 256), 256, 0, stream>>>(emb, Xbf, (int)NX);
  repack_wT   <<<(int)(NW / 256), 256, 0, stream>>>(Wq, WqT);
  repack_wT   <<<(int)(NW / 256), 256, 0, stream>>>(Wk, WkT);
  repack_wT   <<<(int)(NW / 256), 256, 0, stream>>>(Wv, WvT);
  transpose_wT<<<(int)(NW / 256), 256, 0, stream>>>(Wo, WoT);

  dim3 gg(8, 32);   // N/128, M/128
  gemm_bf16_wmma<<<gg, 256, 0, stream>>>(Xbf, WqT, bq, nullptr, Qm,  1);
  gemm_bf16_wmma<<<gg, 256, 0, stream>>>(Xbf, WkT, bk, nullptr, Km,  1);
  gemm_bf16_wmma<<<gg, 256, 0, stream>>>(Xbf, WvT, bv, nullptr, Vtm, 2);

  attn_wmma<<<dim3(32, 32), 128, 0, stream>>>(Qm, Km, Vtm, Head);

  gemm_bf16_wmma<<<gg, 256, 0, stream>>>(Head, WoT, bo, out, nullptr, 0);
}
